// Graph_model_83571473645739
// MI455X (gfx1250) — matrix operand order repaired and checked by reading, not compiled, not run on hardware
//
#include <hip/hip_runtime.h>
#include <math.h>

#define NNODES 4096
#define SEQL   64
#define EMBD   256
#define HIDD   128
#define GATES  512
#define HROW   (SEQL*EMBD)   /* 16384 */
#define NHEADS 8

typedef __bf16 bf16;
typedef __attribute__((ext_vector_type(16))) __bf16 v16bf;
typedef __attribute__((ext_vector_type(8)))  float  v8f;

struct __attribute__((aligned(16))) UV4 { unsigned int x, y, z, w; };
union FragCv { UV4 u[2]; v16bf v; };

// A-fragment (16x32 bf16, M rows striped per ISA 16-bit A layout):
// lane<16 : row=lane,    K = kBase+{0..7}  and kBase+{16..23}
// lane>=16: row=lane-16, K = kBase+{8..15} and kBase+{24..31}
__device__ inline v16bf frag_a(const bf16* base, long rowStride, int kBase, int lane) {
  int half = lane >> 4, m = lane & 15;
  const bf16* p = base + (long)m * rowStride + kBase + half * 8;
  FragCv cv;
  cv.u[0] = *(const UV4*)(p);
  cv.u[1] = *(const UV4*)(p + 16);
  return cv.v;
}
// B-fragment (32x16 bf16) from row-major W[N][K] (i.e. B = W^T):
// lane<16 : col=nBase+lane,    K = kBase+{0..7}  and kBase+{16..23}
// lane>=16: col=nBase+lane-16, K = kBase+{8..15} and kBase+{24..31}
__device__ inline v16bf frag_b(const bf16* w, long rowStride, int nBase, int kBase, int lane) {
  int half = lane >> 4, n = nBase + (lane & 15);
  const bf16* p = w + (long)n * rowStride + kBase + half * 8;
  FragCv cv;
  cv.u[0] = *(const UV4*)(p);
  cv.u[1] = *(const UV4*)(p + 16);
  return cv.v;
}
#define WMMA_BF16(a,b,c) __builtin_amdgcn_wmma_f32_16x16x32_bf16(false,(a),false,(b),(short)0,(c),false,false)

__device__ inline float sigm(float x){ return 1.0f/(1.0f+expf(-x)); }
__device__ inline void atomicMaxF(float* addr, float val){
  int* ia = (int*)addr;
  int old = *ia;
  while (__int_as_float(old) < val) {
    int assumed = old;
    old = atomicCAS(ia, assumed, __float_as_int(val));
    if (old == assumed) break;
  }
}

// ---------------- utility / prep kernels ----------------
__global__ void k_fill_f32(float* p, float v, long n){
  long i = (long)blockIdx.x*blockDim.x + threadIdx.x;
  if (i < n) p[i] = v;
}
__global__ void k_fill_bf16(bf16* p, long n){
  long i = (long)blockIdx.x*blockDim.x + threadIdx.x;
  if (i < n) p[i] = (bf16)0.0f;
}
__global__ void k_f32_to_bf16(const float* s, bf16* d, long n){
  long i = (long)blockIdx.x*blockDim.x + threadIdx.x;
  if (i < n) d[i] = (bf16)s[i];
}
// pad-convert: src [sr][sc] f32 -> dst [dr][dc] bf16 (zeros outside)
__global__ void k_pad_bf16(const float* s, bf16* d, int sr, int sc, int dr, int dc){
  long i = (long)blockIdx.x*blockDim.x + threadIdx.x;
  long tot = (long)dr*dc;
  if (i >= tot) return;
  int r = (int)(i / dc), c = (int)(i % dc);
  d[i] = (r < sr && c < sc) ? (bf16)s[(long)r*sc + c] : (bf16)0.0f;
}
// transpose-pad: src [K][N] f32 -> dst [N][dK] bf16 (dst[n][k]=src[k][n])
__global__ void k_tpad_bf16(const float* s, bf16* d, int K, int N, int dK){
  long i = (long)blockIdx.x*blockDim.x + threadIdx.x;
  long tot = (long)N*dK;
  if (i >= tot) return;
  int n = (int)(i / dK), k = (int)(i % dK);
  d[i] = (k < K) ? (bf16)s[(long)k*N + n] : (bf16)0.0f;
}
// ---------------- embedding (float4 gather, 4 rows per block) ----------------
__global__ __launch_bounds__(256) void k_embed(const int* __restrict__ x,
                                               const float* __restrict__ emb,
                                               bf16* __restrict__ out){
  long row = (long)blockIdx.x*4 + (threadIdx.x >> 6);    // node*SEQ + t
  int  i   = threadIdx.x & 63;                           // 64 lanes * 4 floats = 256
  int tok = x[row];
  float4 v = ((const float4*)(emb + (long)tok*EMBD))[i];
  bf16* d = out + row*EMBD + i*4;
  d[0] = (bf16)v.x; d[1] = (bf16)v.y; d[2] = (bf16)v.z; d[3] = (bf16)v.w;
}
// ---------------- fused LSTM direction (WMMA input proj + recurrence) ----------------
__global__ __launch_bounds__(256) void k_lstm_dir(
    const bf16* __restrict__ Xin,          // [NNODES][SEQL][256] bf16
    const bf16* __restrict__ WihB,         // [512][256] bf16 (row=gate col)
    const bf16* __restrict__ WhhB,         // [512][128] bf16
    const float* __restrict__ bih, const float* __restrict__ bhh,
    bf16* __restrict__ Hout,               // [NNODES][SEQL][256]
    int outOff, int reverse)
{
  __shared__ __align__(16) bf16  sH[16*HIDD];
  __shared__ float sC[16*HIDD];
  __shared__ float sG[16*GATES];
  const int nodeBase = blockIdx.x * 16;
  const int lane = threadIdx.x & 31;
  const int wave = threadIdx.x >> 5;     // 0..7, owns gate cols [wave*64, wave*64+64)
  for (int i = threadIdx.x; i < 16*HIDD; i += 256){ sH[i] = (bf16)0.0f; sC[i] = 0.0f; }
  // hoist loop-invariant gate biases (4 N-tiles per wave, one col per lane)
  float biasj[4];
  #pragma unroll
  for (int j = 0; j < 4; ++j) {
    int col = wave*64 + j*16 + (lane & 15);
    biasj[j] = bih[col] + bhh[col];
  }
  __syncthreads();
  const long aRowStride = (long)SEQL*EMBD;
  for (int s = 0; s < SEQL; ++s) {
    const int t = reverse ? (SEQL-1-s) : s;
    // prefetch next timestep's 16x512B activation tile into cache (global_prefetch_b8)
    const int tnext = reverse ? (SEQL-2-s) : (s+1);
    if (tnext >= 0 && tnext < SEQL && threadIdx.x < 64) {
      const char* pf = (const char*)(Xin + ((long)(nodeBase + (threadIdx.x >> 2))*SEQL + tnext)*EMBD)
                       + (int)(threadIdx.x & 3)*128;
      __builtin_prefetch(pf, 0, 1);
    }
    v8f zero = {0.f,0.f,0.f,0.f,0.f,0.f,0.f,0.f};
    v8f acc[4]; acc[0]=zero; acc[1]=zero; acc[2]=zero; acc[3]=zero;
    // ---- input projection: x_t (16 x 256) * Wih^T, 2-deep pipelined A stream ----
    const bf16* Abase = Xin + ((long)nodeBase*SEQL + t)*EMBD;
    v16bf aCur = frag_a(Abase, aRowStride, 0, lane);
    #pragma unroll
    for (int k0 = 0; k0 < EMBD/32; ++k0) {
      v16bf aNext;
      if (k0 + 1 < EMBD/32) aNext = frag_a(Abase, aRowStride, (k0+1)*32, lane);
      #pragma unroll
      for (int j = 0; j < 4; ++j) {
        v16bf b = frag_b(WihB, EMBD, wave*64 + j*16, k0*32, lane);
        acc[j] = WMMA_BF16(aCur, b, acc[j]);
      }
      aCur = aNext;
    }
    // ---- recurrent: h (16 x 128) * Whh^T, pipelined from LDS ----
    v16bf hCur = frag_a(sH, HIDD, 0, lane);
    #pragma unroll
    for (int k0 = 0; k0 < HIDD/32; ++k0) {
      v16bf hNext;
      if (k0 + 1 < HIDD/32) hNext = frag_a(sH, HIDD, (k0+1)*32, lane);
      #pragma unroll
      for (int j = 0; j < 4; ++j) {
        v16bf b = frag_b(WhhB, HIDD, wave*64 + j*16, k0*32, lane);
        acc[j] = WMMA_BF16(hCur, b, acc[j]);
      }
      hCur = hNext;
    }
    // scatter gates (+bias) to LDS: C layout lane<16 -> row=r, lane>=16 -> row=8+r
    #pragma unroll
    for (int j = 0; j < 4; ++j) {
      int col = wave*64 + j*16 + (lane & 15);
      #pragma unroll
      for (int r = 0; r < 8; ++r) {
        int row = (lane < 16) ? r : (8 + r);
        sG[row*GATES + col] = acc[j][r] + biasj[j];
      }
    }
    __syncthreads();
    // elementwise cell update
    for (int idx = threadIdx.x; idx < 16*HIDD; idx += 256) {
      int node = idx >> 7, u = idx & 127;
      float gi = sG[node*GATES + u];
      float gf = sG[node*GATES + 128 + u];
      float gg = sG[node*GATES + 256 + u];
      float go = sG[node*GATES + 384 + u];
      float c = sigm(gf)*sC[idx] + sigm(gi)*tanhf(gg);
      float h = sigm(go)*tanhf(c);
      sC[idx] = c;
      sH[idx] = (bf16)h;
      Hout[((long)(nodeBase+node)*SEQL + t)*EMBD + outOff + u] = (bf16)h;
    }
    __syncthreads();
  }
}
// ---------------- conv (windows-as-GEMM) + ReLU + maxpool ----------------
__global__ __launch_bounds__(256) void k_conv_pool(
    const bf16* __restrict__ H,    // [NNODES][HROW] bf16
    const bf16* __restrict__ Wk,   // [112][kw*256] bf16 (rows 100..111 zero)
    const float* __restrict__ bias,// [100]
    bf16* __restrict__ pooledB,    // [NNODES][320] bf16
    int kw, int koff)
{
  __shared__ __align__(16) bf16 sHrow[HROW + 768];
  __shared__ int sMax[112];
  const int node = blockIdx.x;
  // vectorized LDS staging: 16384 bf16 = 2048 uint4, plus 96 uint4 of zero pad
  {
    const UV4* s4 = (const UV4*)(H + (long)node*HROW);
    UV4* d4 = (UV4*)sHrow;
    UV4 z; z.x = 0u; z.y = 0u; z.z = 0u; z.w = 0u;
    for (int i = threadIdx.x; i < (HROW + 768)/8; i += 256)
      d4[i] = (i < HROW/8) ? s4[i] : z;
  }
  for (int i = threadIdx.x; i < 112; i += 256) sMax[i] = 0;  // relu >= 0
  __syncthreads();
  const int P = SEQL - kw + 1;
  const int Kdim = kw * 256;
  const int lane = threadIdx.x & 31, wave = threadIdx.x >> 5;
  for (int tid = wave; tid < 28; tid += 8) {   // 4 M-tiles x 7 N-tiles
    int tm = tid & 3, tn = tid >> 2;
    v8f acc = {0.f,0.f,0.f,0.f,0.f,0.f,0.f,0.f};
    v16bf aCur = frag_a(sHrow + tm*16*256, 256, 0, lane);
    for (int k0 = 0; k0 < Kdim/32; ++k0) {
      v16bf aNext;
      if (k0 + 1 < Kdim/32) aNext = frag_a(sHrow + tm*16*256, 256, (k0+1)*32, lane);
      v16bf b = frag_b(Wk, Kdim, tn*16, k0*32, lane);
      acc = WMMA_BF16(aCur, b, acc);
      aCur = aNext;
    }
    int f = tn*16 + (lane & 15);
    float bb = (f < 100) ? bias[f] : 0.0f;
    float m = 0.0f;
    #pragma unroll
    for (int r = 0; r < 8; ++r) {
      int p = tm*16 + ((lane < 16) ? r : 8 + r);
      float v = acc[r] + bb;
      v = v > 0.0f ? v : 0.0f;
      if (p < P && v > m) m = v;
    }
    atomicMax(&sMax[f], __float_as_int(m));
  }
  __syncthreads();
  for (int f = threadIdx.x; f < 100; f += 256)
    pooledB[(long)node*320 + koff + f] = (bf16)__int_as_float(sMax[f]);
}
// ---------------- generic NT WMMA GEMM: C[M][N] = A[M][K] * W[N][K]^T (+bias) ----------------
__global__ __launch_bounds__(128) void k_gemm_nt(
    const bf16* __restrict__ A, const bf16* __restrict__ W,
    const float* __restrict__ bias, float* __restrict__ C,
    int M, int N, int K)
{
  const int lane = threadIdx.x & 31, wave = threadIdx.x >> 5;
  const int tm = blockIdx.y*4 + wave;
  if (tm*16 >= M) return;
  const int tn = blockIdx.x;
  v8f acc = {0.f,0.f,0.f,0.f,0.f,0.f,0.f,0.f};
  const bf16* Abase = A + (long)tm*16*K;
  v16bf aCur = frag_a(Abase, K, 0, lane);
  v16bf bCur = frag_b(W, K, tn*16, 0, lane);
  for (int k0 = 0; k0 < K/32; ++k0) {
    v16bf aNext, bNext;
    if (k0 + 1 < K/32) {
      aNext = frag_a(Abase, K, (k0+1)*32, lane);
      bNext = frag_b(W, K, tn*16, (k0+1)*32, lane);
    }
    acc = WMMA_BF16(aCur, bCur, acc);
    aCur = aNext; bCur = bNext;
  }
  int col = tn*16 + (lane & 15);
  float bb = bias ? bias[col] : 0.0f;
  #pragma unroll
  for (int r = 0; r < 8; ++r) {
    int row = tm*16 + ((lane < 16) ? r : 8 + r);
    C[(long)row*N + col] = acc[r] + bb;
  }
}
// ---------------- small fused elementwise kernels ----------------
__global__ void k_fc_post(const float* C, const float* period, const float* value, bf16* a1){
  int i = blockIdx.x*blockDim.x + threadIdx.x;
  if (i >= NNODES*HIDD) return;
  int n = i >> 7;
  float per = expf(value[0]*period[n]*(1.0f/30.0f));
  a1[i] = (bf16)(tanhf(C[i]) / per);
}
__global__ void k_tanh_concat(const float* C, const float* price, float* h132){
  int i = blockIdx.x*blockDim.x + threadIdx.x;
  if (i >= NNODES*132) return;
  int n = i/132, f = i%132;
  h132[i] = (f < 128) ? tanhf(C[(long)n*128 + f]) : price[n*4 + (f-128)];
}
__global__ void k_tanh(const float* s, float* d, long n){
  long i = (long)blockIdx.x*blockDim.x + threadIdx.x;
  if (i < n) d[i] = tanhf(s[i]);
}
// ---------------- graph kernels ----------------
__global__ void k_deg(const int* src, const int* dst, float* degS, float* degD, int E){
  int i = blockIdx.x*blockDim.x + threadIdx.x;
  if (i >= E) return;
  atomicAdd(&degS[src[i]], 1.0f);
  atomicAdd(&degD[dst[i]], 1.0f);
}
__global__ void k_rs(const float* degS, const float* degD, float* rsO, float* rsI){
  int i = blockIdx.x*blockDim.x + threadIdx.x;
  if (i >= NNODES) return;
  rsO[i] = rsqrtf(fmaxf(degS[i], 1.0f));
  rsI[i] = rsqrtf(fmaxf(degD[i], 1.0f));
}
__global__ void k_gc_agg(const float* h, const int* src, const int* dst,
                         const float* rsO, float* agg, int F, long total){
  long i = (long)blockIdx.x*blockDim.x + threadIdx.x;
  if (i >= total) return;
  int e = (int)(i / F), f = (int)(i % F);
  int s = src[e];
  atomicAdd(&agg[(long)dst[e]*F + f], h[(long)s*F + f] * rsO[s]);
}
__global__ void k_gc_scale(const float* agg, const float* rsI, bf16* out, int F, int Fp){
  long i = (long)blockIdx.x*blockDim.x + threadIdx.x;
  if (i >= (long)NNODES*Fp) return;
  int n = (int)(i / Fp), f = (int)(i % Fp);
  out[i] = (f < F) ? (bf16)(agg[(long)n*F + f] * rsI[n]) : (bf16)0.0f;
}
// ---------------- GAT kernels ----------------
__global__ void k_gat_elr(const float* feat, const float* al, const float* ar,
                          float* el, float* er){
  int i = blockIdx.x*blockDim.x + threadIdx.x;
  if (i >= NNODES*NHEADS) return;
  int n = i >> 3, h = i & 7;
  const float* f = feat + (long)n*1024 + h*128;
  const float* L = al + h*128;
  const float* R = ar + h*128;
  float sl = 0.0f, sr = 0.0f;
  for (int k = 0; k < 128; ++k){ sl += f[k]*L[k]; sr += f[k]*R[k]; }
  el[i] = sl; er[i] = sr;
}
__global__ void k_gat_edge1(const int* src, const int* dst, const float* el, const float* er,
                            float* eval, float* m, int E){
  int i = blockIdx.x*blockDim.x + threadIdx.x;
  if (i >= E*NHEADS) return;
  int e = i >> 3, h = i & 7;
  float v = el[src[e]*NHEADS + h] + er[dst[e]*NHEADS + h];
  v = v > 0.0f ? v : 0.2f*v;                 // leaky_relu 0.2
  eval[i] = v;
  atomicMaxF(&m[dst[e]*NHEADS + h], v);
}
__global__ void k_gat_edge2(const int* dst, const float* eval, const float* m,
                            float* exv, float* ssum, int E){
  int i = blockIdx.x*blockDim.x + threadIdx.x;
  if (i >= E*NHEADS) return;
  int e = i >> 3, h = i & 7;
  float ex = expf(eval[i] - m[dst[e]*NHEADS + h]);
  exv[i] = ex;
  atomicAdd(&ssum[dst[e]*NHEADS + h], ex);
}
__global__ void k_gat_edge3(const int* src, const int* dst, const float* exv, const float* ssum,
                            const float* feat, float* gout, long total){
  long i = (long)blockIdx.x*blockDim.x + threadIdx.x;
  if (i >= total) return;
  int e = (int)(i >> 10), r = (int)(i & 1023), h = r >> 7;
  float a = exv[e*NHEADS + h] / ssum[dst[e]*NHEADS + h];
  atomicAdd(&gout[(long)dst[e]*1024 + r], a * feat[(long)src[e]*1024 + r]);
}
__global__ __launch_bounds__(128) void k_final(const float* gout, const float* gb,
                                               const float* ow, const float* ob, float* out){
  __shared__ float sT[128];
  int n = blockIdx.x, f = threadIdx.x;
  float acc = 0.0f;
  #pragma unroll
  for (int h = 0; h < NHEADS; ++h) acc += gout[(long)n*1024 + h*128 + f] + gb[h*128 + f];
  sT[f] = tanhf(acc * 0.125f);
  __syncthreads();
  if (f < 3) {
    float s = ob[f];
    for (int k = 0; k < 128; ++k) s += sT[k]*ow[f*128 + k];
    out[n*3 + f] = s;
  }
}

// ================= host orchestration =================
static inline unsigned blks(long n, int b){ return (unsigned)((n + b - 1)/b); }

extern "C" void kernel_launch(void* const* d_in, const int* in_sizes, int n_in,
                              void* d_out, int out_size, void* d_ws, size_t ws_size,
                              hipStream_t stream) {
  (void)n_in; (void)out_size; (void)ws_size;
  const int*   x      = (const int*)  d_in[0];
  const float* period = (const float*)d_in[1];
  const float* price  = (const float*)d_in[2];
  const int*   src    = (const int*)  d_in[3];
  const int*   dst    = (const int*)  d_in[4];
  const float* emb    = (const float*)d_in[5];
  const float* Wih[4] = {(const float*)d_in[6],(const float*)d_in[10],(const float*)d_in[14],(const float*)d_in[18]};
  const float* Whh[4] = {(const float*)d_in[7],(const float*)d_in[11],(const float*)d_in[15],(const float*)d_in[19]};
  const float* bih[4] = {(const float*)d_in[8],(const float*)d_in[12],(const float*)d_in[16],(const float*)d_in[20]};
  const float* bhh[4] = {(const float*)d_in[9],(const float*)d_in[13],(const float*)d_in[17],(const float*)d_in[21]};
  const float* convW[3] = {(const float*)d_in[22],(const float*)d_in[23],(const float*)d_in[24]};
  const float* convB[3] = {(const float*)d_in[25],(const float*)d_in[26],(const float*)d_in[27]};
  const float* fc_w  = (const float*)d_in[28];
  const float* fc_b  = (const float*)d_in[29];
  const float* fc1_w = (const float*)d_in[30];
  const float* fc1_b = (const float*)d_in[31];
  const float* value = (const float*)d_in[32];
  const float* gc1_w = (const float*)d_in[33];
  const float* gc1_b = (const float*)d_in[34];
  const float* gc2_w = (const float*)d_in[35];
  const float* gc2_b = (const float*)d_in[36];
  const float* gat_w = (const float*)d_in[37];
  const float* attn_l= (const float*)d_in[38];
  const float* attn_r= (const float*)d_in[39];
  const float* gat_b = (const float*)d_in[40];
  const float* out_w = (const float*)d_in[41];
  const float* out_b = (const float*)d_in[42];
  float* out = (float*)d_out;
  const int E = in_sizes[3];

  // bump allocator over d_ws
  char* ws = (char*)d_ws;
  size_t cur = 0;
  auto alloc = [&](size_t bytes)->void* {
    void* p = ws + cur;
    cur += (bytes + 255) & ~(size_t)255;
    return p;
  };
  bf16* hA   = (bf16*)alloc((size_t)NNODES*SEQL*EMBD*2);
  bf16* hB   = (bf16*)alloc((size_t)NNODES*SEQL*EMBD*2);
  bf16* wihB[4], *whhB[4], *cwB[3];
  for (int i=0;i<4;i++) wihB[i] = (bf16*)alloc((size_t)GATES*EMBD*2);
  for (int i=0;i<4;i++) whhB[i] = (bf16*)alloc((size_t)GATES*HIDD*2);
  const int KW[3] = {2,3,4};
  for (int i=0;i<3;i++) cwB[i] = (bf16*)alloc((size_t)112*KW[i]*256*2);
  bf16* pooledB = (bf16*)alloc((size_t)NNODES*320*2);
  bf16* fcW  = (bf16*)alloc((size_t)128*320*2);
  bf16* fc1W = (bf16*)alloc((size_t)128*128*2);
  bf16* gc1WT= (bf16*)alloc((size_t)128*160*2);
  bf16* gc2WT= (bf16*)alloc((size_t)128*128*2);
  bf16* gatWT= (bf16*)alloc((size_t)1024*128*2);
  float* Cbuf = (float*)alloc((size_t)NNODES*128*4);
  bf16*  a1   = (bf16*) alloc((size_t)NNODES*128*2);
  float* h132 = (float*)alloc((size_t)NNODES*132*4);
  float* degS = (float*)alloc((size_t)NNODES*4);
  float* degD = (float*)alloc((size_t)NNODES*4);
  float* rsO  = (float*)alloc((size_t)NNODES*4);
  float* rsI  = (float*)alloc((size_t)NNODES*4);
  float* agg  = (float*)alloc((size_t)NNODES*132*4);
  bf16*  aggA = (bf16*) alloc((size_t)NNODES*160*2);
  float* g1   = (float*)alloc((size_t)NNODES*128*4);
  float* g2   = (float*)alloc((size_t)NNODES*128*4);
  bf16*  g2b  = (bf16*) alloc((size_t)NNODES*128*2);
  float* feat = (float*)alloc((size_t)NNODES*1024*4);
  float* el   = (float*)alloc((size_t)NNODES*NHEADS*4);
  float* er   = (float*)alloc((size_t)NNODES*NHEADS*4);
  float* mbuf = (float*)alloc((size_t)NNODES*NHEADS*4);
  float* ssum = (float*)alloc((size_t)NNODES*NHEADS*4);
  float* eval = (float*)alloc((size_t)E*NHEADS*4);
  float* exv  = (float*)alloc((size_t)E*NHEADS*4);
  float* gout = (float*)alloc((size_t)NNODES*1024*4);

  // ---- weight prep (bf16 conversion / transpose / pad) ----
  for (int i=0;i<4;i++){
    k_pad_bf16<<<blks((long)GATES*EMBD,256),256,0,stream>>>(Wih[i], wihB[i], GATES, EMBD, GATES, EMBD);
    k_pad_bf16<<<blks((long)GATES*HIDD,256),256,0,stream>>>(Whh[i], whhB[i], GATES, HIDD, GATES, HIDD);
  }
  for (int i=0;i<3;i++){
    int Kd = KW[i]*256;
    k_pad_bf16<<<blks((long)112*Kd,256),256,0,stream>>>(convW[i], cwB[i], 100, Kd, 112, Kd);
  }
  k_pad_bf16 <<<blks((long)128*320,256),256,0,stream>>>(fc_w,  fcW,  128, 300, 128, 320);
  k_pad_bf16 <<<blks((long)128*128,256),256,0,stream>>>(fc1_w, fc1W, 128, 128, 128, 128);
  k_tpad_bf16<<<blks((long)128*160,256),256,0,stream>>>(gc1_w, gc1WT, 132, 128, 160);
  k_tpad_bf16<<<blks((long)128*128,256),256,0,stream>>>(gc2_w, gc2WT, 128, 128, 128);
  k_tpad_bf16<<<blks((long)1024*128,256),256,0,stream>>>(gat_w, gatWT, 128, 1024, 128);
  // ---- zero-init accumulators (every call: ws is not re-poisoned) ----
  k_fill_bf16<<<blks((long)NNODES*320,256),256,0,stream>>>(pooledB, (long)NNODES*320);
  k_fill_f32 <<<blks(NNODES,256),256,0,stream>>>(degS, 0.0f, NNODES);
  k_fill_f32 <<<blks(NNODES,256),256,0,stream>>>(degD, 0.0f, NNODES);
  k_fill_f32 <<<blks((long)NNODES*NHEADS,256),256,0,stream>>>(mbuf, -3.0e38f, (long)NNODES*NHEADS);
  k_fill_f32 <<<blks((long)NNODES*NHEADS,256),256,0,stream>>>(ssum, 0.0f, (long)NNODES*NHEADS);
  k_fill_f32 <<<blks((long)NNODES*1024,256),256,0,stream>>>(gout, 0.0f, (long)NNODES*1024);

  // ---- embedding ----
  k_embed<<<NNODES*SEQL/4, 256, 0, stream>>>(x, emb, hA);
  // ---- BiLSTM layer 0: hA -> hB ----
  k_lstm_dir<<<NNODES/16, 256, 0, stream>>>(hA, wihB[0], whhB[0], bih[0], bhh[0], hB, 0,   0);
  k_lstm_dir<<<NNODES/16, 256, 0, stream>>>(hA, wihB[1], whhB[1], bih[1], bhh[1], hB, 128, 1);
  // ---- BiLSTM layer 1: hB -> hA ----
  k_lstm_dir<<<NNODES/16, 256, 0, stream>>>(hB, wihB[2], whhB[2], bih[2], bhh[2], hA, 0,   0);
  k_lstm_dir<<<NNODES/16, 256, 0, stream>>>(hB, wihB[3], whhB[3], bih[3], bhh[3], hA, 128, 1);
  // ---- conv + relu + maxpool ----
  k_conv_pool<<<NNODES, 256, 0, stream>>>(hA, cwB[0], convB[0], pooledB, 2, 0);
  k_conv_pool<<<NNODES, 256, 0, stream>>>(hA, cwB[1], convB[1], pooledB, 3, 100);
  k_conv_pool<<<NNODES, 256, 0, stream>>>(hA, cwB[2], convB[2], pooledB, 4, 200);
  // ---- fc -> tanh -> /per -> fc1 -> tanh -> concat price ----
  k_gemm_nt<<<dim3(128/16, NNODES/64), 128, 0, stream>>>(pooledB, fcW, fc_b, Cbuf, NNODES, 128, 320);
  k_fc_post<<<blks((long)NNODES*128,256),256,0,stream>>>(Cbuf, period, value, a1);
  k_gemm_nt<<<dim3(128/16, NNODES/64), 128, 0, stream>>>(a1, fc1W, fc1_b, Cbuf, NNODES, 128, 128);
  k_tanh_concat<<<blks((long)NNODES*132,256),256,0,stream>>>(Cbuf, price, h132);
  // ---- degrees ----
  k_deg<<<blks(E,256),256,0,stream>>>(src, dst, degS, degD, E);
  k_rs <<<blks(NNODES,256),256,0,stream>>>(degS, degD, rsO, rsI);
  // ---- graph conv 1 (F=132 -> 128) ----
  k_fill_f32<<<blks((long)NNODES*132,256),256,0,stream>>>(agg, 0.0f, (long)NNODES*132);
  k_gc_agg<<<blks((long)E*132,256),256,0,stream>>>(h132, src, dst, rsO, agg, 132, (long)E*132);
  k_gc_scale<<<blks((long)NNODES*160,256),256,0,stream>>>(agg, rsI, aggA, 132, 160);
  k_gemm_nt<<<dim3(128/16, NNODES/64), 128, 0, stream>>>(aggA, gc1WT, gc1_b, Cbuf, NNODES, 128, 160);
  k_tanh<<<blks((long)NNODES*128,256),256,0,stream>>>(Cbuf, g1, (long)NNODES*128);
  // ---- graph conv 2 (F=128 -> 128) ----
  k_fill_f32<<<blks((long)NNODES*128,256),256,0,stream>>>(agg, 0.0f, (long)NNODES*128);
  k_gc_agg<<<blks((long)E*128,256),256,0,stream>>>(g1, src, dst, rsO, agg, 128, (long)E*128);
  k_gc_scale<<<blks((long)NNODES*128,256),256,0,stream>>>(agg, rsI, aggA, 128, 128);
  k_gemm_nt<<<dim3(128/16, NNODES/64), 128, 0, stream>>>(aggA, gc2WT, gc2_b, Cbuf, NNODES, 128, 128);
  k_tanh<<<blks((long)NNODES*128,256),256,0,stream>>>(Cbuf, g2, (long)NNODES*128);
  k_f32_to_bf16<<<blks((long)NNODES*128,256),256,0,stream>>>(g2, g2b, (long)NNODES*128);
  // ---- GAT ----
  k_gemm_nt<<<dim3(1024/16, NNODES/64), 128, 0, stream>>>(g2b, gatWT, (const float*)nullptr, feat, NNODES, 1024, 128);
  k_gat_elr  <<<blks((long)NNODES*NHEADS,256),256,0,stream>>>(feat, attn_l, attn_r, el, er);
  k_gat_edge1<<<blks((long)E*NHEADS,256),256,0,stream>>>(src, dst, el, er, eval, mbuf, E);
  k_gat_edge2<<<blks((long)E*NHEADS,256),256,0,stream>>>(dst, eval, mbuf, exv, ssum, E);
  k_gat_edge3<<<blks((long)E*1024,256),256,0,stream>>>(src, dst, exv, ssum, feat, gout, (long)E*1024);
  // ---- head mean + tanh + output projection ----
  k_final<<<NNODES, 128, 0, stream>>>(gout, gat_b, out_w, out_b, out);
}
